// NTXentLoss_3985729651337
// MI455X (gfx1250) — matrix operand order repaired and checked by reading, not compiled, not run on hardware
//
#include <hip/hip_runtime.h>
#include <math.h>

typedef _Float16 f16;
typedef __attribute__((ext_vector_type(8)))  _Float16 v8h;
typedef __attribute__((ext_vector_type(16))) _Float16 v16h;
typedef __attribute__((ext_vector_type(8)))  float    v8f;

#define B_ROWS 4096
#define N2     8192
#define DIM    128
#define NTILES (N2 / 16)   // 512 column tiles
#define NRBLK  (N2 / 32)   // 256 row-blocks (32 rows each)
#define NSPLIT 8           // column splits per row-block
#define CT_PER_SPLIT (NTILES / NSPLIT)  // 64 column tiles per wave
#define NWAVES (NRBLK * NSPLIT)         // 2048 waves

// exp(sim / T) = exp2(sim * (1/T) * log2(e));  T = 0.5
#define EXP2_SCALE 2.8853900817779268f   // 2 * log2(e)

// raw v_exp_f32 (hardware exp2)
__device__ __forceinline__ float fast_exp2(float x) {
    return __builtin_amdgcn_exp2f(x);
}

// Workspace layout (bytes):
//   [0      .. 128)    : float blockPart[32]        (row-lse block partials)
//   [2048   .. 18432)  : float posPartial[4096]
//   [65536  .. 327680) : float sPart[8][8192]       (per-split row sum-exp)
//   [327680 .. +2MB)   : f16 rn[8192][128]          (normalized rows)
#define WS_BLK_OFF 0
#define WS_POS_OFF 2048
#define WS_SP_OFF  65536
#define WS_RN_OFF  327680

// ---------------------------------------------------------------------------
// Kernel 1: row L2-normalize, emit f16 copy. One wave per row.
// ---------------------------------------------------------------------------
__global__ __launch_bounds__(256) void ntx_normalize(const float* __restrict__ zi,
                                                     const float* __restrict__ zj,
                                                     f16* __restrict__ rn)
{
    int wave = (blockIdx.x * blockDim.x + threadIdx.x) >> 5;
    int lane = threadIdx.x & 31;
    if (wave >= N2) return;
    const float* src = (wave < B_ROWS) ? (zi + (size_t)wave * DIM)
                                       : (zj + (size_t)(wave - B_ROWS) * DIM);
    float v[4];
    float ss = 0.f;
#pragma unroll
    for (int k = 0; k < 4; ++k) {
        v[k] = src[lane + 32 * k];
        ss += v[k] * v[k];
    }
#pragma unroll
    for (int m = 16; m >= 1; m >>= 1) ss += __shfl_xor(ss, m, 32);
    float scale = 1.0f / fmaxf(sqrtf(ss), 1e-8f);
    f16* dst = rn + (size_t)wave * DIM;
#pragma unroll
    for (int k = 0; k < 4; ++k) dst[lane + 32 * k] = (f16)(v[k] * scale);
}

// ---------------------------------------------------------------------------
// B fragment loader: 32x16 (KxN) f16 tile, col N = lane&15,
//   lane<16 -> K[32c+0..7, 32c+16..23], lane>=16 -> K[32c+8..15, 32c+24..31] per chunk c.
// ---------------------------------------------------------------------------
__device__ __forceinline__ void loadB(const f16* __restrict__ rn, int colBase,
                                      int l16, bool hi, v16h bf[4])
{
    const f16* brow = rn + (size_t)(colBase + l16) * DIM;
#pragma unroll
    for (int c = 0; c < 4; ++c) {
        int off = 32 * c + (hi ? 8 : 0);
        v8h b0 = *(const v8h*)(brow + off);
        v8h b1 = *(const v8h*)(brow + off + 16);
        v16h b;
#pragma unroll
        for (int e = 0; e < 8; ++e) { b[e] = b0[e]; b[e + 8] = b1[e]; }
        bf[c] = b;
    }
}

// ---------------------------------------------------------------------------
// Kernel 2: tiled sim = rn @ rn^T via WMMA, unmasked sum-exp (logits bounded
// in [-2,2], no max needed). Diagonal removed post-loop via recomputed f16
// self-dot (the diagonal value up to f32 reduction-order rounding).
// Wave (rowBlk, split) owns 32 rows x 64 column tiles; double-buffered B.
// ---------------------------------------------------------------------------
__global__ __launch_bounds__(256, 1) void ntx_simlse(const f16* __restrict__ rn,
                                                     float* __restrict__ sPart)
{
    int wave    = (blockIdx.x * blockDim.x + threadIdx.x) >> 5;
    int lane    = threadIdx.x & 31;
    int rowBlk  = wave >> 3;          // 0..255
    int split   = wave & 7;           // 0..7
    int rowBase0 = rowBlk * 32;
    int rowBase1 = rowBase0 + 16;
    bool hi      = lane >= 16;
    int  l16     = lane & 15;
    const int rA = hi ? 8 : 0;

    // ---- A fragments for both row tiles (CDNA5 16-bit A layout) ----
    v16h a0[4], a1[4];
#pragma unroll
    for (int t = 0; t < 2; ++t) {
        const f16* arow = rn + (size_t)((t ? rowBase1 : rowBase0) + l16) * DIM;
#pragma unroll
        for (int c = 0; c < 4; ++c) {
            int off = 32 * c + (hi ? 8 : 0);
            v8h lo = *(const v8h*)(arow + off);
            v8h hh = *(const v8h*)(arow + off + 16);
            v16h a;
#pragma unroll
            for (int e = 0; e < 8; ++e) { a[e] = lo[e]; a[e + 8] = hh[e]; }
            (t ? a1 : a0)[c] = a;
        }
    }

    float s0[8], s1[8];
#pragma unroll
    for (int r = 0; r < 8; ++r) { s0[r] = 0.f; s1[r] = 0.f; }

    const int ctBegin = split * CT_PER_SPLIT;
    const int ctEnd   = ctBegin + CT_PER_SPLIT;

    v16h bbufA[4], bbufB[4];
    loadB(rn, ctBegin * 16, l16, hi, bbufA);

    for (int ct = ctBegin; ct < ctEnd; ct += 2) {
        // ---- prefetch tile ct+1 while computing ct ----
        loadB(rn, (ct + 1) * 16, l16, hi, bbufB);
        {
            v8f d0 = {}, d1 = {};
#pragma unroll
            for (int c = 0; c < 4; ++c) {
                d0 = __builtin_amdgcn_wmma_f32_16x16x32_f16(
                        false, a0[c], false, bbufA[c], (short)0, d0, false, false);
                d1 = __builtin_amdgcn_wmma_f32_16x16x32_f16(
                        false, a1[c], false, bbufA[c], (short)0, d1, false, false);
            }
#pragma unroll
            for (int r = 0; r < 8; ++r) {
                s0[r] += fast_exp2(d0[r] * EXP2_SCALE);
                s1[r] += fast_exp2(d1[r] * EXP2_SCALE);
            }
        }
        // ---- prefetch tile ct+2 while computing ct+1 ----
        if (ct + 2 < ctEnd) loadB(rn, (ct + 2) * 16, l16, hi, bbufA);
        {
            v8f d0 = {}, d1 = {};
#pragma unroll
            for (int c = 0; c < 4; ++c) {
                d0 = __builtin_amdgcn_wmma_f32_16x16x32_f16(
                        false, a0[c], false, bbufB[c], (short)0, d0, false, false);
                d1 = __builtin_amdgcn_wmma_f32_16x16x32_f16(
                        false, a1[c], false, bbufB[c], (short)0, d1, false, false);
            }
#pragma unroll
            for (int r = 0; r < 8; ++r) {
                s0[r] += fast_exp2(d0[r] * EXP2_SCALE);
                s1[r] += fast_exp2(d1[r] * EXP2_SCALE);
            }
        }
    }

    // ---- remove diagonal contribution (only the split that computed it) ----
    // Row rowBase0+l16's full 128-K data is split between lanes l16 and l16+16:
    // square-sum local A elements, add the partner half via one shuffle.
    bool ownsDiag = (split == (rowBlk >> 5));
    float sd0 = 0.f, sd1 = 0.f;
#pragma unroll
    for (int c = 0; c < 4; ++c) {
#pragma unroll
        for (int e = 0; e < 16; ++e) {
            float x0 = (float)a0[c][e];
            float x1 = (float)a1[c][e];
            sd0 += x0 * x0;
            sd1 += x1 * x1;
        }
    }
    sd0 += __shfl_xor(sd0, 16, 32);   // self-dot of row rowBase0 + l16
    sd1 += __shfl_xor(sd1, 16, 32);   // self-dot of row rowBase1 + l16
    float e0 = fast_exp2(sd0 * EXP2_SCALE);
    float e1 = fast_exp2(sd1 * EXP2_SCALE);
    // diag of row (base + l16) lives in slot r = l16 - rA of this lane
#pragma unroll
    for (int r = 0; r < 8; ++r) {
        bool mine = ownsDiag && (l16 == r + rA);
        s0[r] -= mine ? e0 : 0.f;
        s1[r] -= mine ? e1 : 0.f;
    }

    // combine sums across the 16 lanes sharing each row
#pragma unroll
    for (int r = 0; r < 8; ++r) {
#pragma unroll
        for (int mk = 1; mk <= 8; mk <<= 1) {
            s0[r] += __shfl_xor(s0[r], mk, 16);
            s1[r] += __shfl_xor(s1[r], mk, 16);
        }
    }
    // lanes 0 / 16 publish rows {base+rA+0..7}; deterministic unique slots
    if (l16 == 0) {
        float* dst = sPart + (size_t)split * N2;
#pragma unroll
        for (int r = 0; r < 8; ++r) {
            dst[rowBase0 + rA + r] = s0[r];
            dst[rowBase1 + rA + r] = s1[r];
        }
    }
}

// ---------------------------------------------------------------------------
// Kernel 3: per-row lse = log(sum over 8 splits), tree-reduced per block.
// ---------------------------------------------------------------------------
__global__ __launch_bounds__(256) void ntx_rowlse(const float* __restrict__ sPart,
                                                  float* __restrict__ blockPart)
{
    __shared__ float sm[256];
    int t = threadIdx.x;
    int row = blockIdx.x * 256 + t;
    float S = 0.f;
#pragma unroll
    for (int sp = 0; sp < NSPLIT; ++sp) S += sPart[(size_t)sp * N2 + row];
    sm[t] = __logf(S);
    __syncthreads();
    for (int stride = 128; stride >= 1; stride >>= 1) {
        if (t < stride) sm[t] += sm[t + stride];
        __syncthreads();
    }
    if (t == 0) blockPart[blockIdx.x] = sm[0];
}

// ---------------------------------------------------------------------------
// Kernel 4: positives pos[i] = dot(rn[i], rn[i+B]); one wave per i < B.
// ---------------------------------------------------------------------------
__global__ __launch_bounds__(256) void ntx_pos(const f16* __restrict__ rn,
                                               float* __restrict__ posPartial)
{
    int wave = (blockIdx.x * blockDim.x + threadIdx.x) >> 5;
    int lane = threadIdx.x & 31;
    if (wave >= B_ROWS) return;
    const f16* a = rn + (size_t)wave * DIM;
    const f16* b = rn + (size_t)(wave + B_ROWS) * DIM;
    float dot = 0.f;
#pragma unroll
    for (int k = 0; k < 4; ++k)
        dot += (float)a[lane + 32 * k] * (float)b[lane + 32 * k];
#pragma unroll
    for (int m = 16; m >= 1; m >>= 1) dot += __shfl_xor(dot, m, 32);
    if (lane == 0) posPartial[wave] = dot;
}

// ---------------------------------------------------------------------------
// Kernel 5: deterministic final reduction.
// loss = (sum(lse) - (1/T)*sum(pos over 2B rows)) / 2B = (sum(lse) - 4*S)/8192
// ---------------------------------------------------------------------------
__global__ __launch_bounds__(256) void ntx_finalize(const float* __restrict__ blockPart,
                                                    const float* __restrict__ posP,
                                                    float* __restrict__ out)
{
    __shared__ float sm[256];
    int t = threadIdx.x;
    float a = (t < 32) ? blockPart[t] : 0.f;
    float b = 0.f;
    for (int i = t; i < B_ROWS; i += 256) b += posP[i];
    sm[t] = a - 4.0f * b;
    __syncthreads();
    for (int stride = 128; stride >= 1; stride >>= 1) {
        if (t < stride) sm[t] += sm[t + stride];
        __syncthreads();
    }
    if (t == 0) out[0] = sm[0] / (float)N2;
}

// ---------------------------------------------------------------------------
extern "C" void kernel_launch(void* const* d_in, const int* in_sizes, int n_in,
                              void* d_out, int out_size, void* d_ws, size_t ws_size,
                              hipStream_t stream)
{
    (void)in_sizes; (void)n_in; (void)out_size; (void)ws_size;
    const float* zi = (const float*)d_in[0];
    const float* zj = (const float*)d_in[1];
    float* out = (float*)d_out;

    char* ws = (char*)d_ws;
    float* blockPart  = (float*)(ws + WS_BLK_OFF);
    float* posPartial = (float*)(ws + WS_POS_OFF);
    float* sPart      = (float*)(ws + WS_SP_OFF);
    f16*   rn         = (f16*)  (ws + WS_RN_OFF);

    ntx_normalize<<<(N2 * 32) / 256, 256, 0, stream>>>(zi, zj, rn);
    // 2048 waves (256 row-blocks x 8 column splits), 8 waves/block -> 256 blocks
    ntx_simlse<<<NWAVES / 8, 256, 0, stream>>>(rn, sPart);
    ntx_rowlse<<<N2 / 256, 256, 0, stream>>>(sPart, blockPart);
    ntx_pos<<<(B_ROWS * 32) / 256, 256, 0, stream>>>(rn, posPartial);
    ntx_finalize<<<1, 256, 0, stream>>>(blockPart, posPartial, out);
}
